// QuantizedExpert_42176578847199
// MI455X (gfx1250) — matrix operand order repaired and checked by reading, not compiled, not run on hardware
//
#include <hip/hip_runtime.h>

// ---------------------------------------------------------------------------
// Trellis-quantized expert FFN for MI455X (gfx1250, wave32, WMMA).
//   out = silu(x Wg^T) * (x Wu^T) @ Wd^T,  W = lut[walks] * sl ⊗ sr * 0.02
// sign_r folded into activations; weights dequantized on the fly into
// v_wmma_f32_16x16x32_f16 B-fragments (16 consecutive int32 codes per lane,
// LUT gathered from LDS). walks traffic (~704MB) is the roofline; WMMA f16
// keeps the 22.6 GFLOP far below the 23.3 TB/s memory floor (~30us).
// ---------------------------------------------------------------------------

typedef __attribute__((ext_vector_type(16))) _Float16 v16h;
typedef __attribute__((ext_vector_type(8)))  _Float16 v8h;
typedef __attribute__((ext_vector_type(8)))  float    v8f;
typedef __attribute__((ext_vector_type(4)))  int      v4i;

#define B_DIM   64
#define H_DIM   4096
#define I_DIM   14336
#define NLUT    4096
#define W_SCALE 0.02f

// ---- WMMA wrapper ---------------------------------------------------------
__device__ __forceinline__ v8f wmma_f16(v16h a, v16h b, v8f c) {
  // 8 args: (neg_a, A, neg_b, B, c_mod, C, reuse_a, reuse_b)
  return __builtin_amdgcn_wmma_f32_16x16x32_f16(false, a, false, b,
                                                (short)0, c, false, false);
}

// ---- A-fragment (16x32 f16) from a row-major f16 matrix -------------------
// Contiguous-K layout used by this code (B matches): lane L holds row M = L&15; lanes<16 get K {0..15},
// lanes>=16 get K {16..31}. Each half is 32B contiguous in memory.
__device__ __forceinline__ v16h load_a_frag(const _Float16* __restrict__ base,
                                            int ld, int m0, int k0, int lane) {
  const int m  = m0 + (lane & 15);
  const int kb = (lane >> 4) << 4;                 // 0 or 16
  const _Float16* p = base + (size_t)m * ld + (k0 + kb);
  v8h lo = *(const v8h*)p;                         // K = k0+kb   .. +7
  v8h hi = *(const v8h*)(p + 8);                   // K = k0+kb+8 .. +15
  v16h r;
#pragma unroll
  for (int j = 0; j < 8; ++j) { r[j] = lo[j]; r[8 + j] = hi[j]; }
  return r;
}

// ---- B-fragment (32x16 f16) dequantized from trellis codes ----------------
// Contiguous-K layout used by this code (A matches): lane L holds column N = L&15; lanes<16 K=0..15, lanes>=16
// K=16..31 -> 16 *consecutive* int32 codes from one walks row per lane.
__device__ __forceinline__ v16h dequant_b_frag(const int* __restrict__ walks,
                                               int ld, int n0, int k0, int lane,
                                               const float* __restrict__ slut) {
  const int n    = n0 + (lane & 15);
  const int koff = lane & 16;                      // 0 or 16
  const int* p   = walks + (size_t)n * ld + (k0 + koff);
  v4i c0 = *(const v4i*)(p);
  v4i c1 = *(const v4i*)(p + 4);
  v4i c2 = *(const v4i*)(p + 8);
  v4i c3 = *(const v4i*)(p + 12);
  v16h r;
#pragma unroll
  for (int e = 0; e < 4; ++e) {                    // LDS gathers + cvt f32->f16
    r[0  + e] = (_Float16)slut[c0[e]];
    r[4  + e] = (_Float16)slut[c1[e]];
    r[8  + e] = (_Float16)slut[c2[e]];
    r[12 + e] = (_Float16)slut[c3[e]];
  }
  return r;
}

// ---- Kernel 0: fold sign_r into activations, convert to f16 ---------------
__global__ void __launch_bounds__(256)
k0_prep(const float* __restrict__ x,
        const float* __restrict__ gsr, const float* __restrict__ usr,
        _Float16* __restrict__ xg, _Float16* __restrict__ xu) {
  int idx = blockIdx.x * blockDim.x + threadIdx.x;
  if (idx >= B_DIM * H_DIM) return;
  int k   = idx & (H_DIM - 1);
  float v = x[idx];
  xg[idx] = (_Float16)(v * gsr[k]);
  xu[idx] = (_Float16)(v * usr[k]);
}

// ---- Kernel 1: hidden = silu(scale*gsl*(xg Wg'^T)) * scale*usl*(xu Wu'^T) --
// grid = I/64 blocks; 4 waves/block; wave w owns 16 output columns.
__global__ void __launch_bounds__(128)
k1_gate_up(const _Float16* __restrict__ xg, const _Float16* __restrict__ xu,
           const float* __restrict__ lut_g, const float* __restrict__ lut_u,
           const int* __restrict__ gw, const int* __restrict__ uw,
           const float* __restrict__ gsl, const float* __restrict__ usl,
           const float* __restrict__ dsr, _Float16* __restrict__ hid) {
  __shared__ float slut_g[NLUT];
  __shared__ float slut_u[NLUT];
  const int tid = threadIdx.x;
  for (int j = tid; j < NLUT; j += 128) {
    slut_g[j] = lut_g[j];
    slut_u[j] = lut_u[j];
  }
  __syncthreads();

  const int lane = tid & 31;
  const int wave = tid >> 5;
  const int i0   = blockIdx.x * 64 + wave * 16;

  v8f accg[4] = {};
  v8f accu[4] = {};

  for (int kc = 0; kc < H_DIM / 32; ++kc) {
    const int k0 = kc * 32;
    v16h bg = dequant_b_frag(gw, H_DIM, i0, k0, lane, slut_g);
    v16h bu = dequant_b_frag(uw, H_DIM, i0, k0, lane, slut_u);
#pragma unroll
    for (int mt = 0; mt < 4; ++mt) {
      v16h ag = load_a_frag(xg, H_DIM, mt * 16, k0, lane);
      v16h au = load_a_frag(xu, H_DIM, mt * 16, k0, lane);
      accg[mt] = wmma_f16(ag, bg, accg[mt]);
      accu[mt] = wmma_f16(au, bu, accu[mt]);
    }
  }

  // Epilogue: per-lane column i is fixed (C layout: N = lane&15).
  const int   i  = i0 + (lane & 15);
  const float gl = gsl[i] * W_SCALE;
  const float ul = usl[i] * W_SCALE;
  const float dr = dsr[i];                 // fold down_sign_r into hidden
  const int   mrow_off = (lane >> 4) << 3; // lanes>=16 hold M = r+8
#pragma unroll
  for (int mt = 0; mt < 4; ++mt) {
#pragma unroll
    for (int r = 0; r < 8; ++r) {
      const int b = mt * 16 + r + mrow_off;
      float g = accg[mt][r] * gl;
      float u = accu[mt][r] * ul;
      float s = g / (1.0f + __expf(-g));   // silu
      hid[(size_t)b * I_DIM + i] = (_Float16)(s * u * dr);
    }
  }
}

// ---- Kernel 2: out = hidden @ Wd'^T, scale*dsl epilogue -------------------
// grid = H/16 blocks; 4 waves split K=14336 four ways; LDS tree reduction
// (deterministic, no atomics).
__global__ void __launch_bounds__(128)
k2_down(const _Float16* __restrict__ hid, const float* __restrict__ lut_d,
        const int* __restrict__ dw, const float* __restrict__ dsl,
        float* __restrict__ out) {
  __shared__ float slut[NLUT];
  __shared__ float sred[4][4][8][32];
  const int tid = threadIdx.x;
  for (int j = tid; j < NLUT; j += 128) slut[j] = lut_d[j];
  __syncthreads();

  const int lane = tid & 31;
  const int wave = tid >> 5;
  const int h0   = blockIdx.x * 16;

  v8f acc[4] = {};
  const int kc_per_wave = (I_DIM / 32) / 4;        // 112
  const int kc0         = wave * kc_per_wave;
  for (int kc = kc0; kc < kc0 + kc_per_wave; ++kc) {
    const int k0 = kc * 32;
    v16h b = dequant_b_frag(dw, I_DIM, h0, k0, lane, slut);
#pragma unroll
    for (int mt = 0; mt < 4; ++mt) {
      v16h a = load_a_frag(hid, I_DIM, mt * 16, k0, lane);
      acc[mt] = wmma_f16(a, b, acc[mt]);
    }
  }

#pragma unroll
  for (int mt = 0; mt < 4; ++mt)
#pragma unroll
    for (int r = 0; r < 8; ++r)
      sred[wave][mt][r][lane] = acc[mt][r];
  __syncthreads();

  if (wave == 0) {
    const int   h  = h0 + (lane & 15);
    const float dl = dsl[h] * W_SCALE;
    const int   mrow_off = (lane >> 4) << 3;
#pragma unroll
    for (int mt = 0; mt < 4; ++mt) {
#pragma unroll
      for (int r = 0; r < 8; ++r) {
        float s = sred[0][mt][r][lane] + sred[1][mt][r][lane] +
                  sred[2][mt][r][lane] + sred[3][mt][r][lane];
        const int b = mt * 16 + r + mrow_off;
        out[(size_t)b * H_DIM + h] = s * dl;
      }
    }
  }
}

// ---------------------------------------------------------------------------
extern "C" void kernel_launch(void* const* d_in, const int* in_sizes, int n_in,
                              void* d_out, int out_size, void* d_ws, size_t ws_size,
                              hipStream_t stream) {
  (void)in_sizes; (void)n_in; (void)out_size; (void)ws_size;
  const float* x     = (const float*)d_in[0];
  const float* lut_g = (const float*)d_in[1];
  const float* lut_u = (const float*)d_in[2];
  const float* lut_d = (const float*)d_in[3];
  const int*   gw    = (const int*)d_in[4];
  const int*   uw    = (const int*)d_in[5];
  const int*   dw    = (const int*)d_in[6];
  const float* gsl   = (const float*)d_in[7];
  const float* gsr   = (const float*)d_in[8];
  const float* usl   = (const float*)d_in[9];
  const float* usr   = (const float*)d_in[10];
  const float* dsl   = (const float*)d_in[11];
  const float* dsr   = (const float*)d_in[12];

  char* ws = (char*)d_ws;
  _Float16* xg  = (_Float16*)(ws);                                  // 512 KB
  _Float16* xu  = (_Float16*)(ws + (size_t)512 * 1024);             // 512 KB
  _Float16* hid = (_Float16*)(ws + (size_t)1024 * 1024);            // 1.75 MB
  float* out = (float*)d_out;

  k0_prep<<<(B_DIM * H_DIM + 255) / 256, 256, 0, stream>>>(x, gsr, usr, xg, xu);
  k1_gate_up<<<I_DIM / 64, 128, 0, stream>>>(xg, xu, lut_g, lut_u, gw, uw,
                                             gsl, usl, dsr, hid);
  k2_down<<<H_DIM / 16, 128, 0, stream>>>(hid, lut_d, dw, dsl, out);
}
